// AutoregressiveInstructionHead_42236708388866
// MI455X (gfx1250) — matrix operand order repaired and checked by reading, not compiled, not run on hardware
//
#include <hip/hip_runtime.h>
#include <hip/hip_bf16.h>
#include <math.h>

// Problem constants (from reference)
constexpr int Bc = 256, Dc = 512, Ac = 1024;
constexpr int NOc = 65, NRc = 17, NIc = 2, Ec = 64, Hc = 128;

typedef __attribute__((ext_vector_type(16))) _Float16 v16h;
typedef __attribute__((ext_vector_type(8)))  float    v8f;

__device__ __forceinline__ int clampi(int x, int lo, int hi) {
    return x < lo ? lo : (x > hi ? hi : x);
}

// ---------------------------------------------------------------------------
// Stage 1a: fp_head[b,h] = features[b,:] . W1[h,:D] + b1[h]   (op head: relu)
// ---------------------------------------------------------------------------
__global__ void k_fp(const float* __restrict__ features,
                     const float* __restrict__ op_W1,  const float* __restrict__ op_b1,
                     const float* __restrict__ rs_W1,  const float* __restrict__ rs_b1,
                     const float* __restrict__ rd_W1,  const float* __restrict__ rd_b1,
                     const float* __restrict__ imm_W1, const float* __restrict__ imm_b1,
                     float* __restrict__ h1, float* __restrict__ fp_rs,
                     float* __restrict__ fp_rd, float* __restrict__ fp_imm)
{
    int gid = blockIdx.x * blockDim.x + threadIdx.x;
    if (gid >= 4 * Bc * Hc) return;
    int head = gid / (Bc * Hc);
    int rem  = gid % (Bc * Hc);
    int b = rem / Hc, h = rem % Hc;

    const float* W1; const float* b1; float* outp; int stride;
    switch (head) {
        case 0:  W1 = op_W1;  b1 = op_b1;  outp = h1;     stride = Dc;          break;
        case 1:  W1 = rs_W1;  b1 = rs_b1;  outp = fp_rs;  stride = Dc + Ec;     break;
        case 2:  W1 = rd_W1;  b1 = rd_b1;  outp = fp_rd;  stride = Dc + 2 * Ec; break;
        default: W1 = imm_W1; b1 = imm_b1; outp = fp_imm; stride = Dc + 3 * Ec; break;
    }
    const float* frow = features + (size_t)b * Dc;
    const float* wrow = W1 + (size_t)h * stride;
    float s = b1[h];
    for (int k = 0; k < Dc; ++k) s = fmaf(frow[k], wrow[k], s);
    if (head == 0) s = fmaxf(s, 0.0f);   // op hidden uses relu here
    outp[b * Hc + h] = s;
}

// ---------------------------------------------------------------------------
// Stage 1b: op logits  (B x NO, K=128)
// ---------------------------------------------------------------------------
__global__ void k_logits_o(const float* __restrict__ h1,
                           const float* __restrict__ op_W2,
                           const float* __restrict__ op_b2,
                           float* __restrict__ logits)
{
    int gid = blockIdx.x * blockDim.x + threadIdx.x;
    if (gid >= Bc * NOc) return;
    int b = gid / NOc, v = gid % NOc;
    const float* hrow = h1 + (size_t)b * Hc;
    const float* wrow = op_W2 + (size_t)v * Hc;
    float s = op_b2[v];
    for (int h = 0; h < Hc; ++h) s = fmaf(hrow[h], wrow[h], s);
    logits[b * NOc + v] = s;
}

// Stage 1c: row-wise log_softmax over NO=65 (one thread per batch row)
__global__ void k_logsm_o(const float* __restrict__ logits, float* __restrict__ op_logp)
{
    int b = threadIdx.x;   // launched with 1 block of 256
    const float* row = logits + (size_t)b * NOc;
    float m = -1e30f;
    for (int v = 0; v < NOc; ++v) m = fmaxf(m, row[v]);
    float s = 0.0f;
    for (int v = 0; v < NOc; ++v) s += expf(row[v] - m);
    float ls = logf(s);
    for (int v = 0; v < NOc; ++v) op_logp[b * NOc + v] = row[v] - m - ls;
}

// ---------------------------------------------------------------------------
// Stage 1d: ep_head[a,h] = sum of gathered-embedding . W1-tail products
// ---------------------------------------------------------------------------
__global__ void k_ep(const int* __restrict__ act_o, const int* __restrict__ act_rs,
                     const int* __restrict__ act_rd,
                     const float* __restrict__ opcode_embed, const float* __restrict__ reg_embed,
                     const float* __restrict__ rs_W1, const float* __restrict__ rd_W1,
                     const float* __restrict__ imm_W1,
                     float* __restrict__ ep_rs, float* __restrict__ ep_rd,
                     float* __restrict__ ep_imm)
{
    int gid = blockIdx.x * blockDim.x + threadIdx.x;
    if (gid >= 3 * Ac * Hc) return;
    int head = gid / (Ac * Hc);
    int rem  = gid % (Ac * Hc);
    int a = rem / Hc, h = rem % Hc;

    int o = clampi(act_o[a], 0, NOc - 1);
    const float* oe = opcode_embed + (size_t)o * Ec;
    float s = 0.0f;
    if (head == 0) {
        const float* w = rs_W1 + (size_t)h * (Dc + Ec) + Dc;
        for (int e = 0; e < Ec; ++e) s = fmaf(oe[e], w[e], s);
        ep_rs[a * Hc + h] = s;
    } else if (head == 1) {
        int irs = clampi(act_rs[a], 0, NRc - 1);
        const float* re = reg_embed + (size_t)irs * Ec;
        const float* w  = rd_W1 + (size_t)h * (Dc + 2 * Ec) + Dc;
        for (int e = 0; e < Ec; ++e) s = fmaf(oe[e], w[e], s);
        for (int e = 0; e < Ec; ++e) s = fmaf(re[e], w[Ec + e], s);
        ep_rd[a * Hc + h] = s;
    } else {
        int irs = clampi(act_rs[a], 0, NRc - 1);
        int ird = clampi(act_rd[a], 0, NRc - 1);
        const float* re  = reg_embed + (size_t)irs * Ec;
        const float* rde = reg_embed + (size_t)ird * Ec;
        const float* w   = imm_W1 + (size_t)h * (Dc + 3 * Ec) + Dc;
        for (int e = 0; e < Ec; ++e) s = fmaf(oe[e],  w[e],          s);
        for (int e = 0; e < Ec; ++e) s = fmaf(re[e],  w[Ec + e],     s);
        for (int e = 0; e < Ec; ++e) s = fmaf(rde[e], w[2 * Ec + e], s);
        ep_imm[a * Hc + h] = s;
    }
}

// ---------------------------------------------------------------------------
// Stage 1e: pack W2 heads into per-lane f16 WMMA B-fragments (N padded to 32).
// B (32x16 f16): lane L holds column N = L%16; lanes 0-15 cover K = kk*32+0..15,
// lanes 16-31 cover K = kk*32+16..31, element i -> K = kk*32 + (L/16)*16 + i.
// Frags: f0,f1 = rs nt0/1; f2,f3 = rd; f4 = imm. Each lane's 16 halfs contiguous.
// ---------------------------------------------------------------------------
__global__ void k_bfrag(const float* __restrict__ rs_W2,
                        const float* __restrict__ rd_W2,
                        const float* __restrict__ imm_W2,
                        _Float16* __restrict__ bfrag)
{
    int gid = blockIdx.x * blockDim.x + threadIdx.x;
    if (gid >= 5 * 4 * 32 * 16) return;
    int i  = gid & 15;
    int l  = (gid >> 4) & 31;
    int kk = (gid >> 9) & 3;
    int f  = gid >> 11;

    const float* W2; int V, nt;
    if (f < 2)      { W2 = rs_W2;  V = NRc; nt = f;     }
    else if (f < 4) { W2 = rd_W2;  V = NRc; nt = f - 2; }
    else            { W2 = imm_W2; V = NIc; nt = 0;     }

    int n = (l & 15) + nt * 16;
    int k = kk * 32 + (l >> 4) * 16 + i;
    float v = (n < V) ? W2[n * Hc + k] : 0.0f;
    bfrag[gid] = (_Float16)v;
}

// ---------------------------------------------------------------------------
// Per-head pass, templated on NTILES/V so every fragment index is a
// compile-time constant -> bf/af stay in VGPRs (no scratch spill).
// ---------------------------------------------------------------------------
template <int NTILES, int V>
__device__ __forceinline__ void process_head(
    const float* __restrict__ fp, const float* __restrict__ eprow,
    const _Float16* __restrict__ bfrag, int fbase,
    const float* __restrict__ b2, int idx,
    int wave, int lane, int row, int khalf, int kbase,
    float* __restrict__ ldsw, float* __restrict__ acc)
{
    // B fragments: one contiguous 32B v16h load per lane per fragment
    v16h bf[NTILES][4];
    #pragma unroll
    for (int nt = 0; nt < NTILES; ++nt) {
        #pragma unroll
        for (int kk = 0; kk < 4; ++kk)
            bf[nt][kk] = *(const v16h*)(bfrag +
                ((size_t)((fbase + nt) * 4 + kk) * 32 + lane) * 16);
    }

    #pragma unroll
    for (int t = 0; t < 4; ++t) {
        const int btile = t * 4 + wave;
        const float* fprow = fp + (size_t)(btile * 16 + row) * Hc;

        // A fragments: h = relu(fp + ep), f32 -> f16
        // A layout: lane = row M = lane%16, khalf = lane/16,
        //           elem i -> K = kk*32 + khalf*16 + i
        v16h af[4];
        #pragma unroll
        for (int kk = 0; kk < 4; ++kk) {
            #pragma unroll
            for (int i = 0; i < 16; ++i) {
                int k = kk * 32 + khalf * 16 + i;
                af[kk][i] = (_Float16)fmaxf(fprow[k] + eprow[k], 0.0f);
            }
        }

        #pragma unroll
        for (int nt = 0; nt < NTILES; ++nt) {
            v8f c = {};
            #pragma unroll
            for (int kk = 0; kk < 4; ++kk)
                c = __builtin_amdgcn_wmma_f32_16x16x32_f16(
                        false, af[kk], false, bf[nt][kk], (short)0, c, false, false);
            // D layout: VGPR r -> row r + 8*khalf, col = lane%16
            #pragma unroll
            for (int r = 0; r < 8; ++r)
                ldsw[(r + 8 * khalf) * 32 + nt * 16 + row] = c[r];
        }
        __syncthreads();

        // Per-row log-softmax over V + gather (lanes 16-31 duplicate, benign)
        float m = -1e30f;
        #pragma unroll
        for (int v = 0; v < V; ++v) m = fmaxf(m, ldsw[row * 32 + v] + b2[v]);
        float s = 0.0f;
        #pragma unroll
        for (int v = 0; v < V; ++v) s += expf(ldsw[row * 32 + v] + b2[v] - m);
        acc[t] += (ldsw[row * 32 + idx] + b2[idx]) - m - logf(s);
        __syncthreads();
    }
}

// ---------------------------------------------------------------------------
// Stage 2: one block per action a (1024 blocks), 128 threads = 4 waves;
// each wave owns 4 b-tiles of 16 rows. K=128 via 4 chained
// v_wmma_f32_16x16x32_f16 per N-tile; 80 static WMMA issues per block.
// ---------------------------------------------------------------------------
__global__ __launch_bounds__(128)
__attribute__((amdgpu_waves_per_eu(1)))
void k_heads(
    const float* __restrict__ fp_rs, const float* __restrict__ fp_rd,
    const float* __restrict__ fp_imm,
    const float* __restrict__ ep_rs, const float* __restrict__ ep_rd,
    const float* __restrict__ ep_imm,
    const float* __restrict__ op_logp, const _Float16* __restrict__ bfrag,
    const int* __restrict__ act_o, const int* __restrict__ act_rs,
    const int* __restrict__ act_rd, const int* __restrict__ act_imm,
    const float* __restrict__ rs_b2, const float* __restrict__ rd_b2,
    const float* __restrict__ imm_b2,
    float* __restrict__ out)
{
    const int a     = blockIdx.x;
    const int tid   = threadIdx.x;
    const int wave  = tid >> 5;
    const int lane  = tid & 31;
    const int row   = lane & 15;
    const int khalf = lane >> 4;
    const int kbase = khalf * 8;

    const int o   = clampi(act_o[a],   0, NOc - 1);
    const int irs = clampi(act_rs[a],  0, NRc - 1);
    const int ird = clampi(act_rd[a],  0, NRc - 1);
    const int iim = clampi(act_imm[a], 0, NIc - 1);

    __shared__ float lds_log[4][16 * 32];   // per-wave 16x32 logits staging
    float* ldsw = lds_log[wave];

    float acc[4];
    #pragma unroll
    for (int t = 0; t < 4; ++t) {
        int b = (t * 4 + wave) * 16 + row;
        acc[t] = op_logp[b * NOc + o];      // op-head gathered log-prob
    }

    process_head<2, NRc>(fp_rs,  ep_rs  + (size_t)a * Hc, bfrag, 0, rs_b2,  irs,
                         wave, lane, row, khalf, kbase, ldsw, acc);
    process_head<2, NRc>(fp_rd,  ep_rd  + (size_t)a * Hc, bfrag, 2, rd_b2,  ird,
                         wave, lane, row, khalf, kbase, ldsw, acc);
    process_head<1, NIc>(fp_imm, ep_imm + (size_t)a * Hc, bfrag, 4, imm_b2, iim,
                         wave, lane, row, khalf, kbase, ldsw, acc);

    if (khalf == 0) {
        #pragma unroll
        for (int t = 0; t < 4; ++t) {
            int b = (t * 4 + wave) * 16 + row;
            out[(size_t)b * Ac + a] = acc[t];
        }
    }
}

// ---------------------------------------------------------------------------
// Workspace layout (floats): everything L2-resident (~2.25 MB total)
// ---------------------------------------------------------------------------
extern "C" void kernel_launch(void* const* d_in, const int* in_sizes, int n_in,
                              void* d_out, int out_size, void* d_ws, size_t ws_size,
                              hipStream_t stream)
{
    const float* features     = (const float*)d_in[0];
    const int*   act_o        = (const int*)  d_in[1];
    const int*   act_rs       = (const int*)  d_in[2];
    const int*   act_rd       = (const int*)  d_in[3];
    const int*   act_imm      = (const int*)  d_in[4];
    const float* opcode_embed = (const float*)d_in[5];
    const float* reg_embed    = (const float*)d_in[6];
    const float* op_W1        = (const float*)d_in[7];
    const float* op_b1        = (const float*)d_in[8];
    const float* op_W2        = (const float*)d_in[9];
    const float* op_b2        = (const float*)d_in[10];
    const float* rs_W1        = (const float*)d_in[11];
    const float* rs_b1        = (const float*)d_in[12];
    const float* rs_W2        = (const float*)d_in[13];
    const float* rs_b2        = (const float*)d_in[14];
    const float* rd_W1        = (const float*)d_in[15];
    const float* rd_b1        = (const float*)d_in[16];
    const float* rd_W2        = (const float*)d_in[17];
    const float* rd_b2        = (const float*)d_in[18];
    const float* imm_W1       = (const float*)d_in[19];
    const float* imm_b1       = (const float*)d_in[20];
    const float* imm_W2       = (const float*)d_in[21];
    const float* imm_b2       = (const float*)d_in[22];

    float* out = (float*)d_out;
    float* ws  = (float*)d_ws;

    float*    fp_rs    = ws;                  // B*H = 32768
    float*    fp_rd    = ws + 32768;
    float*    fp_imm   = ws + 65536;
    float*    h1       = ws + 98304;          // B*H
    float*    ep_rs    = ws + 131072;         // A*H = 131072
    float*    ep_rd    = ws + 262144;
    float*    ep_imm   = ws + 393216;
    float*    logits_o = ws + 524288;         // B*NO = 16640
    float*    op_logp  = ws + 540928;         // B*NO
    _Float16* bfrag    = (_Float16*)(ws + 557568);  // 10240 halfs, 32B-aligned

    // Stage 1 (independent small f32 kernels)
    k_fp<<<(4 * Bc * Hc) / 256, 256, 0, stream>>>(
        features, op_W1, op_b1, rs_W1, rs_b1, rd_W1, rd_b1, imm_W1, imm_b1,
        h1, fp_rs, fp_rd, fp_imm);
    k_ep<<<(3 * Ac * Hc) / 256, 256, 0, stream>>>(
        act_o, act_rs, act_rd, opcode_embed, reg_embed,
        rs_W1, rd_W1, imm_W1, ep_rs, ep_rd, ep_imm);
    k_bfrag<<<(5 * 4 * 32 * 16) / 256, 256, 0, stream>>>(rs_W2, rd_W2, imm_W2, bfrag);
    k_logits_o<<<(Bc * NOc + 255) / 256, 256, 0, stream>>>(h1, op_W2, op_b2, logits_o);
    k_logsm_o<<<1, 256, 0, stream>>>(logits_o, op_logp);

    // Stage 2: WMMA contraction + fused softmax/gather/accumulate
    k_heads<<<Ac, 128, 0, stream>>>(
        fp_rs, fp_rd, fp_imm, ep_rs, ep_rd, ep_imm, op_logp, bfrag,
        act_o, act_rs, act_rd, act_imm, rs_b2, rd_b2, imm_b2, out);
}
